// LSTM_82308753260737
// MI455X (gfx1250) — matrix operand order repaired and checked by reading, not compiled, not run on hardware
//
#include <hip/hip_runtime.h>
#include <stdint.h>

// ---------------------------------------------------------------------------
// 2-layer LSTM scan for MI455X (gfx1250, wave32, WMMA).
//
// Mapping: 8 workgroups (one per 16-row batch tile), 512 threads = 16 waves.
// Each wave owns 16 h-columns and computes the i/g/o gate tiles for those
// columns (3 x v_wmma_f32_16x16x32_f16 per K-block, K=512 = [x|h]).
// Recurrent state (c) is fp32 in LDS; activations are f16 in LDS; weights are
// f16 in global (L2-resident, 1.5 MB), re-streamed every step through a
// per-iteration opaque zero offset so the loads stay in-loop (no hoist+spill)
// while keeping global (not flat) address-space provenance.
// Per-step x tiles are DMA'd into LDS with the Tensor Data Mover (double
// buffered, TENSORcnt-synchronized).
//
// Workspace requirement: ~69 MB (Wcat f16 + bias sums + x*fw*tw as f16).
// ---------------------------------------------------------------------------

namespace {
constexpr int Bb = 128, Ss = 1024, Ff = 256, Hh = 256, Ll = 2;
constexpr int Gg = 3 * Hh;     // 768 gate columns
constexpr int Kc = Ff + Hh;    // 512 concatenated K

constexpr size_t OUT_SEQ = (size_t)Bb * Ss * Hh;  // outputs elements
constexpr size_t OUT_HS  = (size_t)Ll * Bb * Hh;  // hs elements

// weight fragment offsets (elements)
constexpr size_t GOFF = (size_t)Hh * Kc;  // gate stride within a layer
constexpr size_t LOFF = (size_t)Gg * Kc;  // layer stride

// workspace layout (bytes)
constexpr size_t WCAT_BYTES = (size_t)Ll * Gg * Kc * 2;  // f16
constexpr size_t WS_WCAT = 0;
constexpr size_t WS_BSUM = WS_WCAT + WCAT_BYTES;          // 2*768 f32
constexpr size_t BSUM_BYTES = (size_t)Ll * Gg * 4;
constexpr size_t WS_XW = WS_BSUM + BSUM_BYTES;            // B*S*F f16
}  // namespace

typedef __attribute__((ext_vector_type(16))) _Float16 v16h;
typedef __attribute__((ext_vector_type(8)))  _Float16 v8h;
typedef __attribute__((ext_vector_type(8)))  float    v8f;
typedef __attribute__((ext_vector_type(4)))  unsigned int v4u;
typedef __attribute__((ext_vector_type(8)))  int      v8i;
typedef __attribute__((ext_vector_type(4)))  int      v4i;

#define MAKE_V16(lo, hi) \
  __builtin_shufflevector((lo), (hi), 0, 1, 2, 3, 4, 5, 6, 7, 8, 9, 10, 11, 12, 13, 14, 15)

// B-fragment: two 16-byte chunks at element offsets `off` and `off`+16 from base w
#define BFRAG(w, off) \
  MAKE_V16(*(const v8h*)((w) + (off)), *(const v8h*)((w) + (off) + 16))

__device__ __forceinline__ float sigmoid_f(float x) {
  return 1.0f / (1.0f + __expf(-x));
}

// --------------------------- prep kernels ----------------------------------

// Wcat[l][n][k] = k<F ? W_ih[l][n][k] : W_hh[l][n][k-F]  (f16, row-major (n,k))
// bsum[l][n]   = b_ih[l][n] + b_hh[l][n]
__global__ void prep_weights_kernel(const float* __restrict__ W_ih,
                                    const float* __restrict__ b_ih,
                                    const float* __restrict__ W_hh,
                                    const float* __restrict__ b_hh,
                                    _Float16* __restrict__ Wcat,
                                    float* __restrict__ bsum) {
  const int total = Ll * Gg * Kc;
  for (int i = blockIdx.x * blockDim.x + threadIdx.x; i < total;
       i += gridDim.x * blockDim.x) {
    const int k = i % Kc;
    const int n = (i / Kc) % Gg;
    const int l = i / (Kc * Gg);
    const float w = (k < Ff) ? W_ih[((size_t)l * Gg + n) * Ff + k]
                             : W_hh[((size_t)l * Gg + n) * Hh + (k - Ff)];
    Wcat[i] = (_Float16)w;
    if (i < Ll * Gg) bsum[i] = b_ih[i] + b_hh[i];
  }
}

// xw[b][s][f] = (f16)(x[b,s,f] * fw[f] * tw[s])
__global__ void prep_x_kernel(const float* __restrict__ x,
                              const float* __restrict__ fw,
                              const float* __restrict__ tw,
                              _Float16* __restrict__ xw) {
  const long long total = (long long)Bb * Ss * Ff;
  for (long long i = blockIdx.x * (long long)blockDim.x + threadIdx.x; i < total;
       i += (long long)gridDim.x * blockDim.x) {
    const int f = (int)(i % Ff);
    const int s = (int)((i / Ff) % Ss);
    xw[i] = (_Float16)(x[i] * fw[f] * tw[s]);
  }
}

// --------------------------- TDM tile load ---------------------------------
// 2-D tile: 16 rows x 256 f16, row stride S*F elements, into LDS at lds_off.
// D# bit layout per CDNA5 ISA (groups 0/1; remaining
// groups NULL content — 2-D tensor). 6-arg builtin on this toolchain.
__device__ __forceinline__ void tdm_load_x_tile(uint32_t lds_off,
                                                const _Float16* gsrc) {
  const uint64_t ga = (uint64_t)(uintptr_t)gsrc;
  v4u g0;
  g0.x = 1u;                                        // count=1 valid descriptor
  g0.y = lds_off;                                   // lds_addr (bytes)
  g0.z = (uint32_t)ga;                              // global_addr[31:0]
  g0.w = (uint32_t)((ga >> 32) & 0x01FFFFFFull)     // global_addr[56:32]
         | (2u << 30);                              // type=2 ("image")
  v8i g1;
  g1[0] = 0x00010000;                               // data_size=1 -> 2 bytes
  g1[1] = (int)(256u << 16);                        // tensor_dim0[15:0] = 256
  g1[2] = (int)(16u << 16);                         // tensor_dim1[15:0] = 16
  g1[3] = (int)(256u << 16);                        // tile_dim0 = 256
  g1[4] = 16;                                       // tile_dim1 = 16
  g1[5] = Ss * Ff;                                  // tensor_dim0_stride[31:0]
  g1[6] = 0;                                        // stride hi / dim1_stride
  g1[7] = 0;
  const v4i z4 = {0, 0, 0, 0};
  const v8i z8 = {0, 0, 0, 0, 0, 0, 0, 0};
  __builtin_amdgcn_tensor_load_to_lds(g0, g1, z4, z4, z8, 0);
}

// --------------------------- main scan kernel ------------------------------

__global__ __launch_bounds__(512, 1) void lstm_scan_kernel(
    const _Float16* __restrict__ xw,    // (B,S,F) f16
    const _Float16* __restrict__ Wcat,  // (L,768,512) f16
    const float* __restrict__ bsum,     // (L,768) f32
    float* __restrict__ out) {          // outputs | hs | cs
  __shared__ __align__(16) _Float16 A0x[2][16 * Ff];  // x_t double buffer
  __shared__ __align__(16) _Float16 A0h[16 * Hh];     // h0_prev (f16)
  __shared__ __align__(16) _Float16 A1x[16 * Hh];     // h0_cur (layer1 input)
  __shared__ __align__(16) _Float16 A1h[16 * Hh];     // h1_prev
  __shared__ __align__(16) float    C0[16 * Hh];      // c0 state (f32)
  __shared__ __align__(16) float    C1[16 * Hh];      // c1 state (f32)

  const int tid    = threadIdx.x;
  const int wid    = tid >> 5;        // wave 0..15 (wave32)
  const int lane   = tid & 31;
  const int halfid = lane >> 4;       // lane half (WMMA layout)
  const int lc     = lane & 15;
  const int bm     = blockIdx.x << 4; // batch tile base

  // zero recurrent state
  for (int i = tid; i < 16 * Hh; i += 512) {
    A0h[i] = (_Float16)0.0f;
    A1h[i] = (_Float16)0.0f;
    C0[i] = 0.0f;
    C1[i] = 0.0f;
  }

  // this lane's h-column and per-gate biases (resident in registers)
  const int nc = (wid << 4) + lc;  // 0..255
  const float bi0 = bsum[nc],      bg0 = bsum[Hh + nc],      bo0 = bsum[2 * Hh + nc];
  const float bi1 = bsum[Gg + nc], bg1 = bsum[Gg + Hh + nc], bo1 = bsum[Gg + 2 * Hh + nc];

  // preload step 0 x-tile via TDM
  if (wid == 0) {
    tdm_load_x_tile((uint32_t)(uintptr_t)&A0x[0][0], xw + (size_t)bm * Ss * Ff);
    __builtin_amdgcn_s_wait_tensorcnt(0);
  }
  __syncthreads();

  for (int s = 0; s < Ss; ++s) {
    const int cb = s & 1, nb = cb ^ 1;

    // Defeat LICM without losing global-addrspace provenance: an opaque
    // zero offset renewed every timestep makes all weight-fragment addresses
    // loop-variant (so the 192 loads stay in-loop, L2-resident streams, no
    // hoist+spill to scratch), while the pointer still derives from the
    // kernarg Wcat (so they lower to global_load_b128, not flat_load).
    uint32_t zoff = 0;
    asm volatile("" : "+v"(zoff));
    // Single per-lane weight base: row n = nc, K sub-block halfid*8.
    // Every fragment in both layers is at a compile-time element offset
    // l*LOFF + gate*GOFF + kk*32 (+16) from this base (max ~1.3 MB, fits the
    // signed 24-bit global_load immediate).
    const _Float16* w = Wcat + (size_t)nc * Kc + (size_t)halfid * 8 + zoff;

    // async: DMA next step's x-tile while this step computes
    if (wid == 0 && s + 1 < Ss) {
      tdm_load_x_tile((uint32_t)(uintptr_t)&A0x[nb][0],
                      xw + ((size_t)bm * Ss + (s + 1)) * Ff);
    }

    // ---------------- layer 0 GEMM: gates = [x_t | h0] @ Wcat0^T + b -------
    v8f aI = {bi0, bi0, bi0, bi0, bi0, bi0, bi0, bi0};
    v8f aG = {bg0, bg0, bg0, bg0, bg0, bg0, bg0, bg0};
    v8f aO = {bo0, bo0, bo0, bo0, bo0, bo0, bo0, bo0};
    {
      const _Float16* ax = &A0x[cb][lc * Ff + halfid * 8];
      const _Float16* ah = &A0h[lc * Hh + halfid * 8];
#pragma unroll
      for (int kk = 0; kk < 16; ++kk) {
        const _Float16* ar = (kk < 8) ? (ax + kk * 32) : (ah + (kk - 8) * 32);
        const v16h a = MAKE_V16(*(const v8h*)ar, *(const v8h*)(ar + 16));
        const size_t ko = (size_t)kk * 32;
        const v16h b0 = BFRAG(w, 0 * GOFF + ko);
        const v16h b1 = BFRAG(w, 1 * GOFF + ko);
        const v16h b2 = BFRAG(w, 2 * GOFF + ko);
        aI = __builtin_amdgcn_wmma_f32_16x16x32_f16(false, a, false, b0, (short)0, aI, false, false);
        aG = __builtin_amdgcn_wmma_f32_16x16x32_f16(false, a, false, b1, (short)0, aG, false, false);
        aO = __builtin_amdgcn_wmma_f32_16x16x32_f16(false, a, false, b2, (short)0, aO, false, false);
      }
    }
    __syncthreads();  // all waves done reading A0h before it is rewritten

    // ---------------- layer 0 pointwise (in-register gates) ----------------
    // D layout: lane holds column nc, rows m = 8*halfid + v.
#pragma unroll
    for (int v = 0; v < 8; ++v) {
      const int m = (halfid << 3) + v;
      const float cp = C0[m * Hh + nc];
      const float fg = sigmoid_f(cp);  // forget = sigmoid(c_prev) per source
      const float cn = fg * cp + sigmoid_f(aI[v]) * tanhf(aG[v]);
      const float h  = sigmoid_f(aO[v]) * tanhf(cn);
      C0[m * Hh + nc] = cn;
      const _Float16 hh = (_Float16)h;
      A1x[m * Hh + nc] = hh;  // layer-1 input this step
      A0h[m * Hh + nc] = hh;  // layer-0 recurrence next step
      if (s == Ss - 1) {
        out[OUT_SEQ + (size_t)(bm + m) * Hh + nc] = h;             // hs[0]
        out[OUT_SEQ + OUT_HS + (size_t)(bm + m) * Hh + nc] = cn;   // cs[0]
      }
    }
    __syncthreads();  // A1x ready for layer-1 GEMM

    // ---------------- layer 1 GEMM: gates = [h0 | h1] @ Wcat1^T + b --------
    v8f bIa = {bi1, bi1, bi1, bi1, bi1, bi1, bi1, bi1};
    v8f bGa = {bg1, bg1, bg1, bg1, bg1, bg1, bg1, bg1};
    v8f bOa = {bo1, bo1, bo1, bo1, bo1, bo1, bo1, bo1};
    {
      const _Float16* ax = &A1x[lc * Hh + halfid * 8];
      const _Float16* ah = &A1h[lc * Hh + halfid * 8];
#pragma unroll
      for (int kk = 0; kk < 16; ++kk) {
        const _Float16* ar = (kk < 8) ? (ax + kk * 32) : (ah + (kk - 8) * 32);
        const v16h a = MAKE_V16(*(const v8h*)ar, *(const v8h*)(ar + 16));
        const size_t ko = (size_t)kk * 32;
        const v16h b0 = BFRAG(w, LOFF + 0 * GOFF + ko);
        const v16h b1 = BFRAG(w, LOFF + 1 * GOFF + ko);
        const v16h b2 = BFRAG(w, LOFF + 2 * GOFF + ko);
        bIa = __builtin_amdgcn_wmma_f32_16x16x32_f16(false, a, false, b0, (short)0, bIa, false, false);
        bGa = __builtin_amdgcn_wmma_f32_16x16x32_f16(false, a, false, b1, (short)0, bGa, false, false);
        bOa = __builtin_amdgcn_wmma_f32_16x16x32_f16(false, a, false, b2, (short)0, bOa, false, false);
      }
    }
    __syncthreads();  // all waves done reading A1h before it is rewritten

    // ---------------- layer 1 pointwise + sequence output ------------------
#pragma unroll
    for (int v = 0; v < 8; ++v) {
      const int m = (halfid << 3) + v;
      const float cp = C1[m * Hh + nc];
      const float fg = sigmoid_f(cp);
      const float cn = fg * cp + sigmoid_f(bIa[v]) * tanhf(bGa[v]);
      const float h  = sigmoid_f(bOa[v]) * tanhf(cn);
      C1[m * Hh + nc] = cn;
      A1h[m * Hh + nc] = (_Float16)h;
      out[((size_t)(bm + m) * Ss + s) * Hh + nc] = h;              // outputs
      if (s == Ss - 1) {
        out[OUT_SEQ + (size_t)Bb * Hh + (size_t)(bm + m) * Hh + nc] = h;            // hs[1]
        out[OUT_SEQ + OUT_HS + (size_t)Bb * Hh + (size_t)(bm + m) * Hh + nc] = cn;  // cs[1]
      }
    }

    if (wid == 0) __builtin_amdgcn_s_wait_tensorcnt(0);  // next x-tile landed
    __syncthreads();
  }
}

// --------------------------- launch ----------------------------------------

extern "C" void kernel_launch(void* const* d_in, const int* in_sizes, int n_in,
                              void* d_out, int out_size, void* d_ws,
                              size_t ws_size, hipStream_t stream) {
  (void)in_sizes; (void)n_in; (void)out_size; (void)ws_size;
  const float* x    = (const float*)d_in[0];
  const float* fw   = (const float*)d_in[1];
  const float* tw   = (const float*)d_in[2];
  const float* W_ih = (const float*)d_in[3];
  const float* b_ih = (const float*)d_in[4];
  const float* W_hh = (const float*)d_in[5];
  const float* b_hh = (const float*)d_in[6];
  float* out = (float*)d_out;

  char* ws = (char*)d_ws;
  _Float16* Wcat = (_Float16*)(ws + WS_WCAT);
  float*    bsum = (float*)(ws + WS_BSUM);
  _Float16* xw   = (_Float16*)(ws + WS_XW);

  prep_weights_kernel<<<1536, 256, 0, stream>>>(W_ih, b_ih, W_hh, b_hh, Wcat, bsum);
  prep_x_kernel<<<8192, 256, 0, stream>>>(x, fw, tw, xw);
  lstm_scan_kernel<<<Bb / 16, 512, 0, stream>>>(xw, Wcat, bsum, out);
}
